// STDPPlasticityRule_56470230008109
// MI455X (gfx1250) — matrix operand order repaired and checked by reading, not compiled, not run on hardware
//
#include <hip/hip_runtime.h>
#include <math.h>

// ---------------------------------------------------------------------------
// STDP plasticity on MI455X (gfx1250, wave32, WMMA, async LDS copies)
//   updates = [x_pre ; pre] (2048x8192 f16)  @  [B0 ; B1]^T (8192x2048 f16)
//   B0 = A_plus*post + A_TRIPLET*post*x_trip ,  B1 = -A_minus*x_post
//   out = clip(W + updates * sqrt(0.1/(activity+1e-6)), -2, 2)
// ---------------------------------------------------------------------------

#define P_DIM 2048
#define Q_DIM 2048
#define B_DIM 16
#define T_DIM 256
#define K_HALF 4096   // B*T
#define K_DIM  8192   // 2 * K_HALF

#define BLK_M 128
#define BLK_N 128
#define BLK_K 64
#define LDS_STRIDE 72   // halfs; 144 bytes = 16B aligned, odd quad-bank step
#define TILE_HALFS (BLK_M * LDS_STRIDE)

typedef _Float16 half8 __attribute__((ext_vector_type(8)));
typedef _Float16 v16h  __attribute__((ext_vector_type(16)));
typedef float    v8f   __attribute__((ext_vector_type(8)));

__device__ __forceinline__ float wave_reduce_add(float v) {
#pragma unroll
  for (int off = 16; off > 0; off >>= 1)
    v += __shfl_xor(v, off, 32);
  return v;
}

// ---- CDNA5 async global->LDS copy (ASYNCcnt-tracked), GVS addressing ------
__device__ __forceinline__ void async_copy_b128(unsigned ldsOff,
                                                unsigned long long gBase,
                                                unsigned gOff) {
#if defined(__HIP_DEVICE_COMPILE__)
  asm volatile("global_load_async_to_lds_b128 %0, %1, %2"
               :: "v"(ldsOff), "v"(gOff), "s"(gBase)
               : "memory");
#else
  (void)ldsOff; (void)gBase; (void)gOff;
#endif
}

__device__ __forceinline__ void wait_async_zero() {
#if defined(__HIP_DEVICE_COMPILE__)
  asm volatile("s_wait_asynccnt 0x0" ::: "memory");
#endif
}

// Low 32 bits of a generic pointer to __shared__ == wave-relative LDS offset
// (ISA 10.2: LDS aperture truncates to addr[31:0]).
__device__ __forceinline__ unsigned lds_off(const void* p) {
  return (unsigned)(unsigned long long)p;
}

// ---------------------------------------------------------------------------
// Kernel 1: pre-synaptic traces -> A matrix [P][K_DIM] f16 (k contiguous)
//   term0 (k in [0,4096)):  x_pre trace (tau_plus)
//   term1 (k in [4096,8192)): raw pre spikes
// ---------------------------------------------------------------------------
__global__ __launch_bounds__(256) void pack_pre(const float* __restrict__ pre,
                                                _Float16* __restrict__ Aw,
                                                float* __restrict__ cnt) {
  const int g = blockIdx.x * blockDim.x + threadIdx.x; // 0 .. B*P-1
  const int b = g / P_DIM;
  const int p = g % P_DIM;
  const float* src = pre + (size_t)(b * P_DIM + p) * T_DIM;
  _Float16* dstX = Aw + (size_t)p * K_DIM + b * T_DIM;
  _Float16* dstS = dstX + K_HALF;

  const float decay = 0.95122945f;  // exp(-DT/TAU_PLUS) = exp(-0.05)
  float carry = 0.f;
  float total = 0.f;
  for (int t0 = 0; t0 < T_DIM; t0 += 8) {
    const float4 s0 = *(const float4*)(src + t0);
    const float4 s1 = *(const float4*)(src + t0 + 4);
    float s[8] = {s0.x, s0.y, s0.z, s0.w, s1.x, s1.y, s1.z, s1.w};
    half8 hx, hs;
#pragma unroll
    for (int i = 0; i < 8; ++i) {
      const float x = decay * carry;   // trace value used at step t
      carry = x + s[i];                // spike added after the update
      total += s[i];
      hx[i] = (_Float16)x;
      hs[i] = (_Float16)s[i];
    }
    *(half8*)(dstX + t0) = hx;
    *(half8*)(dstS + t0) = hs;
  }
  total = wave_reduce_add(total);
  if ((threadIdx.x & 31) == 0) atomicAdd(cnt, total);
}

// ---------------------------------------------------------------------------
// Kernel 2: post-synaptic traces -> Bt matrix [Q][K_DIM] f16 (k contiguous)
//   term0: A_plus*s + A_TRIPLET*s*x_trip      term1: -A_minus*x_post
// ---------------------------------------------------------------------------
__global__ __launch_bounds__(256) void pack_post(const float* __restrict__ post,
                                                 const float* __restrict__ aPlusPtr,
                                                 const float* __restrict__ aMinusPtr,
                                                 _Float16* __restrict__ Bw,
                                                 float* __restrict__ cnt) {
  const int g = blockIdx.x * blockDim.x + threadIdx.x; // 0 .. B*Q-1
  const int b = g / Q_DIM;
  const int q = g % Q_DIM;
  const float aPlus  = aPlusPtr[0];
  const float aMinus = aMinusPtr[0];
  const float aTrip  = 0.0065f;

  const float* src = post + (size_t)(b * Q_DIM + q) * T_DIM;
  _Float16* dst0 = Bw + (size_t)q * K_DIM + b * T_DIM;
  _Float16* dst1 = dst0 + K_HALF;

  const float decayM = 0.95122945f;  // exp(-DT/TAU_MINUS)
  const float decayX = 0.99014786f;  // exp(-DT/TAU_X) = exp(-0.001/0.101)
  float cM = 0.f, cX = 0.f, total = 0.f;
  for (int t0 = 0; t0 < T_DIM; t0 += 8) {
    const float4 s0 = *(const float4*)(src + t0);
    const float4 s1 = *(const float4*)(src + t0 + 4);
    float s[8] = {s0.x, s0.y, s0.z, s0.w, s1.x, s1.y, s1.z, s1.w};
    half8 h0, h1;
#pragma unroll
    for (int i = 0; i < 8; ++i) {
      const float xM = decayM * cM;  cM = xM + s[i];  // x_post
      const float xX = decayX * cX;  cX = xX + s[i];  // x_trip
      total += s[i];
      h0[i] = (_Float16)(aPlus * s[i] + aTrip * s[i] * xX);
      h1[i] = (_Float16)(-aMinus * xM);
    }
    *(half8*)(dst0 + t0) = h0;
    *(half8*)(dst1 + t0) = h1;
  }
  total = wave_reduce_add(total);
  if ((threadIdx.x & 31) == 0) atomicAdd(cnt, total);
}

// ---------------------------------------------------------------------------
// Kernel 3: WMMA GEMM  C[p][q] = sum_k A[p][k]*Bt[q][k]; fused epilogue.
//   block tile 128x128, 8 waves, wave tile 64x32 (4x2 WMMA accumulators)
//   Double-buffered LDS fed by global_load_async_to_lds_b128 (ASYNCcnt).
// ---------------------------------------------------------------------------
__global__ __launch_bounds__(256) void stdp_gemm(const _Float16* __restrict__ Aw,
                                                 const _Float16* __restrict__ Bw,
                                                 const float* __restrict__ W,
                                                 const float* __restrict__ cnt,
                                                 float* __restrict__ out) {
  __shared__ __align__(16) _Float16 As[2][TILE_HALFS];
  __shared__ __align__(16) _Float16 Bs[2][TILE_HALFS];

  const int tid   = threadIdx.x;
  const int lane  = tid & 31;
  const int wave  = tid >> 5;     // 0..7
  const int waveM = wave & 1;     // 2 wave-rows
  const int waveN = wave >> 1;    // 4 wave-cols
  const int blockM = blockIdx.x * BLK_M;
  const int blockN = blockIdx.y * BLK_N;

  const int lr = lane & 15;       // row/col within 16x16 tile
  const int lh = lane >> 4;       // lane half select

  const unsigned asBase = lds_off(&As[0][0]);
  const unsigned bsBase = lds_off(&Bs[0][0]);
  const unsigned long long aG = (unsigned long long)Aw;
  const unsigned long long bG = (unsigned long long)Bw;

  // Issue one stage (32KB total) of async copies: 4x16B for A, 4x16B for B per thread.
  auto issue_stage = [&](int stage, int buf) {
    const int k0 = stage * BLK_K;
#pragma unroll
    for (int i = 0; i < 4; ++i) {
      const int chunk = tid + i * 256;       // 0..1023
      const int r = chunk >> 3;              // 0..127
      const int c = (chunk & 7) << 3;        // 0..56 step 8
      const unsigned ldsA = asBase + (unsigned)((buf * TILE_HALFS + r * LDS_STRIDE + c) * 2);
      const unsigned gA   = (unsigned)(((blockM + r) * K_DIM + k0 + c) * 2);
      async_copy_b128(ldsA, aG, gA);
      const unsigned ldsB = bsBase + (unsigned)((buf * TILE_HALFS + r * LDS_STRIDE + c) * 2);
      const unsigned gB   = (unsigned)(((blockN + r) * K_DIM + k0 + c) * 2);
      async_copy_b128(ldsB, bG, gB);
    }
  };

  v8f acc[4][2] = {};

  constexpr int NST = K_DIM / BLK_K;  // 128 stages
  issue_stage(0, 0);

  for (int s = 0; s < NST; ++s) {
    wait_async_zero();    // this wave's stage-s copies landed in LDS
    __syncthreads();      // all waves' copies landed; all done reading buf (s+1)&1
    if (s + 1 < NST) issue_stage(s + 1, (s + 1) & 1);  // prefetch overlaps compute

    const _Float16* Ab = &As[s & 1][0];
    const _Float16* Bb = &Bs[s & 1][0];

#pragma unroll
    for (int kk = 0; kk < BLK_K; kk += 32) {
      // A fragment (16x32 f16): lane<16 -> K 0..7 then 16..23; lane>=16 -> K 8..15 then 24..31
      v16h af[4];
#pragma unroll
      for (int m = 0; m < 4; ++m) {
        const int row = waveM * 64 + m * 16 + lr;
        const _Float16* base = Ab + row * LDS_STRIDE + kk + lh * 8;
        half8 lo = *(const half8*)(base);
        half8 hi = *(const half8*)(base + 16);
        af[m] = __builtin_shufflevector(lo, hi, 0, 1, 2, 3, 4, 5, 6, 7,
                                                8, 9, 10, 11, 12, 13, 14, 15);
      }
      // B fragment (32x16 f16): lane<16 -> K 0..7 then 16..23; lane>=16 -> K 8..15 then 24..31 (column = lr)
      v16h bf[2];
#pragma unroll
      for (int n = 0; n < 2; ++n) {
        const int row = waveN * 32 + n * 16 + lr;
        const _Float16* base = Bb + row * LDS_STRIDE + kk + lh * 8;
        half8 lo = *(const half8*)(base);
        half8 hi = *(const half8*)(base + 16);
        bf[n] = __builtin_shufflevector(lo, hi, 0, 1, 2, 3, 4, 5, 6, 7,
                                                8, 9, 10, 11, 12, 13, 14, 15);
      }
#pragma unroll
      for (int m = 0; m < 4; ++m)
#pragma unroll
        for (int n = 0; n < 2; ++n)
          acc[m][n] = __builtin_amdgcn_wmma_f32_16x16x32_f16(
              false, af[m], false, bf[n], (short)0, acc[m][n], false, false);
    }
    __syncthreads();  // all waves done reading buf s&1 before stage s+2 overwrites it
  }

  // ---- fused epilogue: homeostatic scaling + clip ----
  const float activity = cnt[0] / (float)(B_DIM * T_DIM);
  const float scale = sqrtf(0.1f / (activity + 1e-6f));

#pragma unroll
  for (int m = 0; m < 4; ++m) {
    const int pBase = blockM + waveM * 64 + m * 16 + lh * 8; // VGPR j -> M = j + 8*lh
#pragma unroll
    for (int n = 0; n < 2; ++n) {
      const int q = blockN + waveN * 32 + n * 16 + lr;       // N = lane%16
#pragma unroll
      for (int j = 0; j < 8; ++j) {
        const size_t idx = (size_t)(pBase + j) * Q_DIM + q;
        float v = W[idx] + acc[m][n][j] * scale;
        v = fminf(fmaxf(v, -2.0f), 2.0f);
        out[idx] = v;
      }
    }
  }
}

// ---------------------------------------------------------------------------
extern "C" void kernel_launch(void* const* d_in, const int* in_sizes, int n_in,
                              void* d_out, int out_size, void* d_ws, size_t ws_size,
                              hipStream_t stream) {
  (void)in_sizes; (void)n_in; (void)out_size; (void)ws_size;
  const float* pre  = (const float*)d_in[0];
  const float* post = (const float*)d_in[1];
  const float* W    = (const float*)d_in[2];
  const float* aP   = (const float*)d_in[3];
  const float* aM   = (const float*)d_in[4];
  float* out = (float*)d_out;

  // Workspace layout: [0,256) counters, then A (32MB f16), then Bt (32MB f16)
  float* cnt = (float*)d_ws;
  _Float16* Aw = (_Float16*)((char*)d_ws + 256);
  _Float16* Bw = Aw + (size_t)P_DIM * K_DIM;

  hipMemsetAsync(d_ws, 0, 256, stream);  // zero spike counter (graph-capturable)

  pack_pre<<<(B_DIM * P_DIM) / 256, 256, 0, stream>>>(pre, Aw, cnt);
  pack_post<<<(B_DIM * Q_DIM) / 256, 256, 0, stream>>>(post, aP, aM, Bw, cnt);

  dim3 grid(P_DIM / BLK_M, Q_DIM / BLK_N);
  stdp_gemm<<<grid, 256, 0, stream>>>(Aw, Bw, W, cnt, out);
}
